// RecurrentEncoder_30674656428410
// MI455X (gfx1250) — matrix operand order repaired and checked by reading, not compiled, not run on hardware
//
#include <hip/hip_runtime.h>
#include <hip/hip_bf16.h>
#include <math.h>

// ---------------------------------------------------------------------------
// Types for CDNA5 WMMA (wave32). v16bf = 8 VGPRs of packed bf16, v8f = 8 VGPRs.
// ---------------------------------------------------------------------------
typedef __bf16 bf16_t;
typedef __attribute__((ext_vector_type(16))) __bf16 v16bf;
typedef __attribute__((ext_vector_type(8)))  __bf16 v8bf;   // one 16-byte chunk
typedef __attribute__((ext_vector_type(8)))  float  v8f;

// Problem constants (from setup_inputs): B=256, D=2048, H=2048, K=64, T=32
#define BB   256
#define DD   2048
#define HH   2048
#define KK   64
#define TT   32
#define G4H  (4 * HH)       // 8192
#define BKT  (BB * KK * TT) // 524288

// ---------------------------------------------------------------------------
// WMMA helpers
// ---------------------------------------------------------------------------
__device__ __forceinline__ v8f wmma_bf16(v16bf a, v16bf b, v8f c) {
  // D = A(16x32 bf16) * B(32x16 bf16) + C(16x16 f32)
  return __builtin_amdgcn_wmma_f32_16x16x32_bf16(
      /*neg_a=*/false, a, /*neg_b=*/false, b,
      /*c_mod=*/(short)0, c, /*reuse_a=*/false, /*reuse_b=*/false);
}

// A fragment: 16x32 bf16 tile, A row-major [M x ld].
// Contiguous k order (same as B): lane L: m = L%16, half = L/16; VGPR0-7 hold K = half*16 .. +16
// -> one contiguous 32B span = two 16B loads per lane.
__device__ __forceinline__ v16bf load_frag_a(const bf16_t* __restrict__ q) {
  v8bf lo = *(const v8bf*)(q);
  v8bf hi = *(const v8bf*)(q + 8);
  v16bf r;
#pragma unroll
  for (int i = 0; i < 8; ++i) { r[i] = lo[i]; r[i + 8] = hi[i]; }
  return r;
}

// B fragment: 32x16 bf16 tile of B[k][n] where B[k][n] = W[n][k], W row-major
// [N x ld]. Contiguous k order (same as A): lane L: n = L%16, half = L/16; VGPR v holds
// K = half*16 + 2v, 2v+1  -> one contiguous 32B span = two 16B loads.
__device__ __forceinline__ v16bf load_frag_b(const bf16_t* __restrict__ q) {
  v8bf lo = *(const v8bf*)(q);
  v8bf hi = *(const v8bf*)(q + 8);
  v16bf r;
#pragma unroll
  for (int i = 0; i < 8; ++i) { r[i] = lo[i]; r[i + 8] = hi[i]; }
  return r;
}

__device__ __forceinline__ float sigmoid_f(float x) {
  return 1.0f / (1.0f + __expf(-x));
}

// ---------------------------------------------------------------------------
// Generic bf16 WMMA GEMM (K = 2048 compile-time): C = A * W^T + bias1 + bias2
// Block = 256 threads (8 waves), block tile 64x64; wave w handles row-tile
// (w&3) and two col-tiles at (w>>2)*2. K loop software-pipelined: fragments
// for k+32 are in flight while the WMMAs for k execute. Prefetch is
// unconditional (strided pointer, no per-iteration branch).
// ---------------------------------------------------------------------------
#define KD 2048
__global__ __launch_bounds__(256) void gemm_bf16_kernel(
    const bf16_t* __restrict__ A, const bf16_t* __restrict__ W,
    const float* __restrict__ bias1, const float* __restrict__ bias2,
    float* __restrict__ C, int N) {
  const int wave = threadIdx.x >> 5;
  const int lane = threadIdx.x & 31;
  const int m = lane & 15, half = lane >> 4;
  const int rowT  = blockIdx.x * 64 + (wave & 3) * 16;
  const int colT0 = blockIdx.y * 64 + (wave >> 2) * 32;

  // lane-resident base pointers (incremented by 32 elems per K step)
  const bf16_t* pa  = A + (size_t)(rowT + m) * KD + half * 16;
  const bf16_t* pb0 = W + (size_t)(colT0 + m) * KD + half * 16;
  const bf16_t* pb1 = W + (size_t)(colT0 + 16 + m) * KD + half * 16;

  v8f acc0 = {}; v8f acc1 = {};
  v16bf a  = load_frag_a(pa);
  v16bf b0 = load_frag_b(pb0);
  v16bf b1 = load_frag_b(pb1);
#pragma unroll 2
  for (int k = 32; k < KD; k += 32) {
    __builtin_prefetch(pa + 96, 0, 1);   // global_prefetch_b8, 2 tiles ahead
    __builtin_prefetch(pb0 + 96, 0, 1);
    pa += 32; pb0 += 32; pb1 += 32;
    v16bf na  = load_frag_a(pa);
    v16bf nb0 = load_frag_b(pb0);
    v16bf nb1 = load_frag_b(pb1);
    acc0 = wmma_bf16(a, b0, acc0);
    acc1 = wmma_bf16(a, b1, acc1);
    a = na; b0 = nb0; b1 = nb1;
  }
  acc0 = wmma_bf16(a, b0, acc0);
  acc1 = wmma_bf16(a, b1, acc1);

  const int mBase = rowT + half * 8;   // C layout: elem e -> M = mBase+e
  const int n0 = colT0 + m;
  float bA = 0.f, bB = 0.f;
  if (bias1) { bA += bias1[n0]; bB += bias1[n0 + 16]; }
  if (bias2) { bA += bias2[n0]; bB += bias2[n0 + 16]; }
#pragma unroll
  for (int e = 0; e < 8; ++e) {
    const size_t row = (size_t)(mBase + e);
    C[row * N + n0]      = acc0[e] + bA;
    C[row * N + n0 + 16] = acc1[e] + bB;
  }
}

// ---------------------------------------------------------------------------
// Fused LSTM step: g = xg + h_in @ W_hh^T, gate activations, c/h update.
// Each wave computes 2 row-tiles x 4 gates (8 accumulators): every B (gate
// weight) fragment is reused twice and 8 independent WMMAs per K-step bury
// the WMMA hazard latency. Pipelined K loop as above.
// Block tile: 128 rows x 32 cols/gate. Grid: (256/128, 2048/32) = (2, 64).
// ---------------------------------------------------------------------------
__global__ __launch_bounds__(256) void lstm_step_kernel(
    const bf16_t* __restrict__ h_in,  // [256 x 2048] bf16 (prev h)
    const bf16_t* __restrict__ Whh,   // [8192 x 2048] bf16
    const float* __restrict__ xg,     // [256 x 8192] f32
    float* __restrict__ c_f,          // [256 x 2048] f32 (in/out)
    float* __restrict__ h_f,          // [256 x 2048] f32 (out)
    bf16_t* __restrict__ h_out) {     // [256 x 2048] bf16 (out, next step A)
  const int wave = threadIdx.x >> 5;
  const int lane = threadIdx.x & 31;
  const int m = lane & 15, half = lane >> 4;
  const int rowT0 = blockIdx.x * 128 + (wave & 3) * 32;
  const int rowT1 = rowT0 + 16;
  const int colT  = blockIdx.y * 32 + (wave >> 2) * 16;  // col within one gate

  const bf16_t* pa0 = h_in + (size_t)(rowT0 + m) * HH + half * 16;
  const bf16_t* pa1 = h_in + (size_t)(rowT1 + m) * HH + half * 16;
  const bf16_t* pb[4];
#pragma unroll
  for (int g = 0; g < 4; ++g)
    pb[g] = Whh + (size_t)(colT + g * HH + m) * HH + half * 16;

  v8f acc[4][2];
#pragma unroll
  for (int g = 0; g < 4; ++g) { acc[g][0] = (v8f){}; acc[g][1] = (v8f){}; }

  v16bf a0 = load_frag_a(pa0);
  v16bf a1 = load_frag_a(pa1);
  v16bf b[4];
#pragma unroll
  for (int g = 0; g < 4; ++g) b[g] = load_frag_b(pb[g]);

#pragma unroll 2
  for (int k = 32; k < HH; k += 32) {
    __builtin_prefetch(pa0 + 96, 0, 1);
    __builtin_prefetch(pb[0] + 96, 0, 1);
    pa0 += 32; pa1 += 32;
    v16bf na0 = load_frag_a(pa0);
    v16bf na1 = load_frag_a(pa1);
    v16bf nb[4];
#pragma unroll
    for (int g = 0; g < 4; ++g) { pb[g] += 32; nb[g] = load_frag_b(pb[g]); }
#pragma unroll
    for (int g = 0; g < 4; ++g) {
      acc[g][0] = wmma_bf16(a0, b[g], acc[g][0]);
      acc[g][1] = wmma_bf16(a1, b[g], acc[g][1]);
    }
    a0 = na0; a1 = na1;
#pragma unroll
    for (int g = 0; g < 4; ++g) b[g] = nb[g];
  }
#pragma unroll
  for (int g = 0; g < 4; ++g) {
    acc[g][0] = wmma_bf16(a0, b[g], acc[g][0]);
    acc[g][1] = wmma_bf16(a1, b[g], acc[g][1]);
  }

  const int n = colT + m;
#pragma unroll
  for (int rt = 0; rt < 2; ++rt) {
    const int mBase = (rt ? rowT1 : rowT0) + half * 8;
#pragma unroll
    for (int e = 0; e < 8; ++e) {
      const int row = mBase + e;
      const float* xr = xg + (size_t)row * G4H;
      const float gi = sigmoid_f(acc[0][rt][e] + xr[n]);
      const float gf = sigmoid_f(acc[1][rt][e] + xr[n + HH]);
      const float gg = tanhf(acc[2][rt][e] + xr[n + 2 * HH]);
      const float go = sigmoid_f(acc[3][rt][e] + xr[n + 3 * HH]);
      const size_t idx = (size_t)row * HH + n;
      const float cn = gf * c_f[idx] + gi * gg;
      const float hn = go * tanhf(cn);
      c_f[idx] = cn;
      h_f[idx] = hn;
      h_out[idx] = (bf16_t)hn;
    }
  }
}

// ---------------------------------------------------------------------------
// Head epilogue: mu/sig/z from fe[256 x 128], scatter into (B,K,T) outputs.
// ---------------------------------------------------------------------------
__global__ void fe_out_kernel(const float* __restrict__ fe,
                              const float* __restrict__ eps,
                              float* __restrict__ out, int t) {
  const int i = blockIdx.x * blockDim.x + threadIdx.x;
  if (i >= BB * KK) return;
  const int b = i >> 6, k = i & 63;
  const float mu = fe[b * 2 * KK + k];
  const float xx = fe[b * 2 * KK + KK + k] - 5.0f;
  const float sig = (xx > 20.0f) ? xx : log1pf(__expf(xx));  // softplus
  const float e = eps[((size_t)t * BB + b) * KK + k];
  const size_t o = ((size_t)b * KK + k) * TT + t;
  out[o] = e * sig + mu;          // zs
  out[BKT + o] = mu;              // mus
  out[2 * BKT + o] = sig;         // sigs
}

// ---------------------------------------------------------------------------
// Utility kernels
// ---------------------------------------------------------------------------
__global__ void f32_to_bf16_x8_kernel(const float* __restrict__ src,
                                      bf16_t* __restrict__ dst, int n8) {
  const int i = blockIdx.x * blockDim.x + threadIdx.x;
  if (i >= n8) return;
  v8f v = *(const v8f*)(src + (size_t)i * 8);
  v8bf o;
#pragma unroll
  for (int j = 0; j < 8; ++j) o[j] = (bf16_t)v[j];
  *(v8bf*)(dst + (size_t)i * 8) = o;
}

__global__ void zero_f32_kernel(float* __restrict__ p, int n) {
  const int i = blockIdx.x * blockDim.x + threadIdx.x;
  if (i < n) p[i] = 0.0f;
}

__global__ void copy_f32_kernel(const float* __restrict__ s,
                                float* __restrict__ d, int n) {
  const int i = blockIdx.x * blockDim.x + threadIdx.x;
  if (i < n) d[i] = s[i];
}

// ---------------------------------------------------------------------------
// Launch
// ---------------------------------------------------------------------------
extern "C" void kernel_launch(void* const* d_in, const int* in_sizes, int n_in,
                              void* d_out, int out_size, void* d_ws, size_t ws_size,
                              hipStream_t stream) {
  (void)in_sizes; (void)n_in; (void)out_size; (void)ws_size;
  const float* x    = (const float*)d_in[0];   // [256, 2048]
  const float* eps  = (const float*)d_in[1];   // [32, 256, 64]
  const float* W_ih = (const float*)d_in[2];   // [8192, 2048]
  const float* W_hh = (const float*)d_in[3];   // [8192, 2048]
  const float* b_ih = (const float*)d_in[4];   // [8192]
  const float* b_hh = (const float*)d_in[5];   // [8192]
  const float* W_fe = (const float*)d_in[6];   // [128, 2048]
  const float* b_fe = (const float*)d_in[7];   // [128]
  float* out = (float*)d_out;

  // Workspace carve-up (all 16B aligned):
  bf16_t* Wih_bf = (bf16_t*)d_ws;                         // 8192*2048 bf16
  bf16_t* Whh_bf = Wih_bf + (size_t)G4H * DD;             // 8192*2048 bf16
  bf16_t* Wfe_bf = Whh_bf + (size_t)G4H * HH;             // 128*2048 bf16
  bf16_t* x_bf   = Wfe_bf + (size_t)2 * KK * HH;          // 256*2048 bf16
  float*  xg     = (float*)(x_bf + (size_t)BB * DD);      // 256*8192 f32
  float*  h_f    = xg + (size_t)BB * G4H;                 // 256*2048 f32
  float*  c_f    = h_f + (size_t)BB * HH;                 // 256*2048 f32
  float*  fe     = c_f + (size_t)BB * HH;                 // 256*128 f32
  bf16_t* h_bf0  = (bf16_t*)(fe + (size_t)BB * 2 * KK);   // 256*2048 bf16
  bf16_t* h_bf1  = h_bf0 + (size_t)BB * HH;               // 256*2048 bf16

  const int TPB = 256;
  // 1) bf16 conversions (weights once, x once), 8 elements/thread
  {
    int n8 = (G4H * DD) / 8;
    f32_to_bf16_x8_kernel<<<(n8 + TPB - 1) / TPB, TPB, 0, stream>>>(W_ih, Wih_bf, n8);
    f32_to_bf16_x8_kernel<<<(n8 + TPB - 1) / TPB, TPB, 0, stream>>>(W_hh, Whh_bf, n8);
    n8 = (2 * KK * HH) / 8;
    f32_to_bf16_x8_kernel<<<(n8 + TPB - 1) / TPB, TPB, 0, stream>>>(W_fe, Wfe_bf, n8);
    n8 = (BB * DD) / 8;
    f32_to_bf16_x8_kernel<<<(n8 + TPB - 1) / TPB, TPB, 0, stream>>>(x, x_bf, n8);
  }
  // 2) init state: c0 = 0, h0(bf16) = 0 (bf16 zero == all-zero bits)
  zero_f32_kernel<<<(BB * HH + TPB - 1) / TPB, TPB, 0, stream>>>(c_f, BB * HH);
  zero_f32_kernel<<<(BB * HH / 2 + TPB - 1) / TPB, TPB, 0, stream>>>((float*)h_bf0, BB * HH / 2);

  // 3) xg = x @ W_ih^T + b_ih + b_hh   (M=256, N=8192, K=2048)
  {
    dim3 grid(BB / 64, G4H / 64);
    gemm_bf16_kernel<<<grid, TPB, 0, stream>>>(x_bf, Wih_bf, b_ih, b_hh, xg, G4H);
  }

  // 4) T recurrent steps (ping-pong h buffers; device-wide sync = launch order)
  for (int t = 0; t < TT; ++t) {
    const bf16_t* hin = (t & 1) ? h_bf1 : h_bf0;
    bf16_t* hout      = (t & 1) ? h_bf0 : h_bf1;
    {
      dim3 grid(BB / 128, HH / 32);  // (2, 64) blocks, 8 waves each
      lstm_step_kernel<<<grid, TPB, 0, stream>>>(hin, Whh_bf, xg, c_f, h_f, hout);
    }
    {
      dim3 grid(BB / 64, (2 * KK) / 64);  // (4, 2): fe = h @ W_fe^T + b_fe
      gemm_bf16_kernel<<<grid, TPB, 0, stream>>>(hout, Wfe_bf, b_fe, nullptr, fe,
                                                 2 * KK);
    }
    fe_out_kernel<<<(BB * KK + TPB - 1) / TPB, TPB, 0, stream>>>(fe, eps, out, t);
  }

  // 5) final (h, c) -> output tail
  copy_f32_kernel<<<(BB * HH + TPB - 1) / TPB, TPB, 0, stream>>>(h_f, out + 3 * BKT, BB * HH);
  copy_f32_kernel<<<(BB * HH + TPB - 1) / TPB, TPB, 0, stream>>>(c_f, out + 3 * BKT + BB * HH, BB * HH);
}
